// LogLinearDeltaProductBlock_89232240542352
// MI455X (gfx1250) — compile-verified, matrix operand order checked by reading, not run on hardware
//
#include <hip/hip_runtime.h>

// ---------------------------------------------------------------------------
// Problem constants (from reference)
// ---------------------------------------------------------------------------
constexpr int Bc  = 2;
constexpr int Tt  = 2048;
constexpr int Dd  = 1024;
constexpr int Hh  = 16;
constexpr int HK  = 64;
constexpr int NH  = 3;
constexpr int CC  = 64;
constexpr int LV  = 16;
constexpr int NCHUNK = Tt / CC;          // 32
constexpr int MTOK = Bc * Tt;            // 4096 token rows

// ---------------------------------------------------------------------------
// Types for WMMA
// ---------------------------------------------------------------------------
typedef __attribute__((ext_vector_type(16))) __bf16    bf16x16;
typedef __attribute__((ext_vector_type(8)))  float     v8f;
typedef __attribute__((ext_vector_type(8)))  unsigned  v8u;

__device__ __forceinline__ __bf16 f2bf(float f) {
  unsigned u = __float_as_uint(f);
  unsigned r = (u + 0x7FFFu + ((u >> 16) & 1u)) >> 16;
  unsigned short s = (unsigned short)r;
  return __builtin_bit_cast(__bf16, s);
}

// LDS byte offset of a __shared__ object (async-to-LDS ops take the raw LDS
// address, not a flat pointer): generic -> addrspace(3) -> int.
__device__ __forceinline__ unsigned lds_off(const void* p) {
  return (unsigned)(unsigned long long)(__attribute__((address_space(3))) const char*)p;
}

// Async global -> LDS copy, 16 bytes per lane. Tracked by ASYNCcnt.
__device__ __forceinline__ void async_ld_b128(unsigned loff, const void* g) {
  asm volatile("global_load_async_to_lds_b128 %0, %1, off"
               :: "v"(loff), "v"(g) : "memory");
}
__device__ __forceinline__ void wait_async0() {
  asm volatile("s_wait_asynccnt 0x0" ::: "memory");
}

// ---------------------------------------------------------------------------
// fp32 -> bf16 conversion (grid-stride)
// ---------------------------------------------------------------------------
__global__ void cvt_bf16_kernel(const float* __restrict__ src, __bf16* __restrict__ dst, size_t n) {
  for (size_t i = (size_t)blockIdx.x * blockDim.x + threadIdx.x; i < n;
       i += (size_t)gridDim.x * blockDim.x)
    dst[i] = f2bf(src[i]);
}

// ---------------------------------------------------------------------------
// WMMA bf16 GEMM:  C[M,N] = A[M,K] * W[N,K]^T   (fp32 accumulate)
// Macro tile 64(M) x 256(N); 8 waves, each wave owns a 32x64 block
// (2x4 of v_wmma_f32_16x16x32_bf16). Double-buffered LDS fed by
// global_load_async_to_lds_b128 (ASYNCcnt), overlapping the next tile's
// transfer with the current tile's WMMAs.
// ---------------------------------------------------------------------------
constexpr int TM = 64, TN = 256, TK = 32;
constexpr int LDT = TK + 8;   // padded LDS row stride (bf16 elements)

__global__ __launch_bounds__(256)
void gemm_bf16_kernel(const __bf16* __restrict__ A, const __bf16* __restrict__ Bw,
                      float* __restrict__ C, int M, int N, int K) {
  __shared__ __bf16 As[2][TM * LDT];   // 2 x 2560  bf16
  __shared__ __bf16 Bs[2][TN * LDT];   // 2 x 10240 bf16

  const int tid  = threadIdx.x;
  const int lane = tid & 31;
  const int wave = tid >> 5;       // 0..7
  const int wm   = wave & 1;       // M-dir wave coord (2 x 32 rows)
  const int wn   = wave >> 1;      // N-dir wave coord (4 x 64 cols)
  const int hlf  = lane >> 4;      // lane half (K +8 select)
  const int l16  = lane & 15;

  const int bm = blockIdx.y * TM;
  const int bn = blockIdx.x * TN;

  v8f acc[2][4] = {};

  // staging assignments
  const int arow = tid >> 2;           // 0..63
  const int akk  = (tid & 3) * 8;      // 0,8,16,24 (bf16 elems; 16B chunks)
  int gbn = bn + tid;                  // B row for this thread (one row each)
  if (gbn >= N) gbn = N - 1;           // clamp: junk cols never stored

  auto stage = [&](int buf, int k0) {
    // A tile: 64x32 bf16 = 4KB = 256 lanes x 16B
    async_ld_b128(lds_off(&As[buf][arow * LDT + akk]),
                  A + (size_t)(bm + arow) * K + k0 + akk);
    // B tile: 256x32 bf16 = 16KB = 256 lanes x 64B (one K-row per lane)
    const __bf16* g = Bw + (size_t)gbn * K + k0;
#pragma unroll
    for (int c4 = 0; c4 < 4; ++c4)
      async_ld_b128(lds_off(&Bs[buf][tid * LDT + c4 * 8]), g + c4 * 8);
  };

  // prologue: stage tile 0
  stage(0, 0);
  wait_async0();
  __syncthreads();

  int cur = 0;
  for (int k0 = 0; k0 < K; k0 += TK) {
    const int nxt = cur ^ 1;
    if (k0 + TK < K) stage(nxt, k0 + TK);   // overlap with compute below

    // ---- build fragments per CDNA5 16-bit A/B lane layout ----
    bf16x16 afrag[2], bfrag[4];
#pragma unroll
    for (int mi = 0; mi < 2; ++mi) {
      const int m = wm * 32 + mi * 16 + l16;
      v8u u;
#pragma unroll
      for (int j = 0; j < 8; ++j) {
        const int kb = 2 * j + 8 * hlf + ((j >= 4) ? 8 : 0);
        u[j] = *(const unsigned*)(&As[cur][m * LDT + kb]);
      }
      afrag[mi] = __builtin_bit_cast(bf16x16, u);
    }
#pragma unroll
    for (int ni = 0; ni < 4; ++ni) {
      const int n = wn * 64 + ni * 16 + l16;
      v8u u;
#pragma unroll
      for (int j = 0; j < 8; ++j) {
        const int kb = 2 * j + 8 * hlf + ((j >= 4) ? 8 : 0);
        u[j] = *(const unsigned*)(&Bs[cur][n * LDT + kb]);
      }
      bfrag[ni] = __builtin_bit_cast(bf16x16, u);
    }

#pragma unroll
    for (int mi = 0; mi < 2; ++mi)
#pragma unroll
      for (int ni = 0; ni < 4; ++ni)
        acc[mi][ni] = __builtin_amdgcn_wmma_f32_16x16x32_bf16(
            false, afrag[mi], false, bfrag[ni], (short)0, acc[mi][ni], false, false);

    wait_async0();     // my async stores into nxt are done
    __syncthreads();   // everyone's are done; cur reads are done
    cur = nxt;
  }

  // ---- epilogue: C/D layout -> global ----
#pragma unroll
  for (int mi = 0; mi < 2; ++mi)
#pragma unroll
    for (int ni = 0; ni < 4; ++ni) {
      const int gnc = bn + wn * 64 + ni * 16 + l16;
#pragma unroll
      for (int r = 0; r < 8; ++r) {
        const int gm = bm + wm * 32 + mi * 16 + r + 8 * hlf;
        if (gm < M && gnc < N) C[(size_t)gm * N + gnc] = acc[mi][ni][r];
      }
    }
}

// ---------------------------------------------------------------------------
// Elementwise activations (in place). mode: 0=silu, 1=sigmoid, 2=log_sigmoid
// ---------------------------------------------------------------------------
__global__ void act_kernel(float* __restrict__ p, size_t n, int mode) {
  size_t i = (size_t)blockIdx.x * blockDim.x + threadIdx.x;
  if (i >= n) return;
  float x = p[i];
  if (mode == 0) {
    p[i] = x / (1.f + __expf(-x));
  } else if (mode == 1) {
    p[i] = 1.f / (1.f + __expf(-x));
  } else {
    p[i] = fminf(x, 0.f) - log1pf(__expf(-fabsf(x)));   // log_sigmoid
  }
}

// ---------------------------------------------------------------------------
// lwc[c][b][h][l] = mean_t softplus(Lp[h][l] * dl[b, c*64+t, h, l])
// ---------------------------------------------------------------------------
__global__ void lwc_kernel(const float* __restrict__ dl, const float* __restrict__ Lp,
                           float* __restrict__ lwc) {
  const int blk = blockIdx.x;
  const int h = blk % Hh;
  const int b = (blk / Hh) % Bc;
  const int c = blk / (Hh * Bc);
  const int tid = threadIdx.x;
  const int l = tid & 15;
  const int grp = tid >> 4;      // 0..3
  const float lp = Lp[h * LV + l];
  float s = 0.f;
  for (int t = grp; t < CC; t += 4) {
    const int tok = c * CC + t;
    const float z = lp * dl[((size_t)b * Tt + tok) * (Hh * LV) + h * LV + l];
    s += fmaxf(z, 0.f) + log1pf(__expf(-fabsf(z)));   // stable softplus
  }
  __shared__ float red[64];
  red[tid] = s;
  __syncthreads();
  if (tid < 16) {
    const float tot = red[tid] + red[tid + 16] + red[tid + 32] + red[tid + 48];
    lwc[(((size_t)c * Bc + b) * Hh + h) * LV + tid] = tot / (float)CC;
  }
}

// ---------------------------------------------------------------------------
// Sequential log-linear DeltaProduct recurrence.
// One 256-thread block per (b,h). S (64x64 fp32) in LDS; 16-level state
// stack in workspace (L2-resident). Thread (vcol=tid%64, rgrp=tid/64) owns
// S rows [16*rgrp, 16*rgrp+16) of column vcol.
// ---------------------------------------------------------------------------
__global__ __launch_bounds__(256)
void recurrence_kernel(const float* __restrict__ q, const float* __restrict__ k,
                       const float* __restrict__ v, const float* __restrict__ bet,
                       const float* __restrict__ g, const float* __restrict__ lwc,
                       float* __restrict__ states, float* __restrict__ outs) {
  const int bh = blockIdx.x;
  const int b = bh / Hh;
  const int h = bh % Hh;
  const int tid = threadIdx.x;
  const int vcol = tid & 63;
  const int row0 = (tid >> 6) * 16;

  __shared__ float S[64 * 64];
  __shared__ float kv[64];
  __shared__ float dv[64];
  __shared__ float red[256];
  __shared__ float lw[LV];
  __shared__ float scl[1];

  // zero this block's state slice (re-done every launch => graph-safe)
  for (int l = 0; l < LV; ++l) {
    float* st = states + ((size_t)(b * LV + l) * Hh + h) * 4096;
    for (int e = tid; e < 4096; e += 256) st[e] = 0.f;
  }
  __syncthreads();

  for (int c = 0; c < NCHUNK; ++c) {
    if (tid < LV) lw[tid] = lwc[(((size_t)c * Bc + b) * Hh + h) * LV + tid];
    __syncthreads();

    // S0 = sum_l lw[l] * states[l]
    for (int i = 0; i < 16; ++i) {
      const int e = (row0 + i) * 64 + vcol;
      float acc = 0.f;
      for (int l = 0; l < LV; ++l)
        acc += lw[l] * states[((size_t)(b * LV + l) * Hh + h) * 4096 + e];
      S[e] = acc;
    }
    __syncthreads();

    for (int t = 0; t < CC; ++t) {
      const int tok = c * CC + t;
      const float scale = __expf(g[((size_t)b * Tt + tok) * Hh + h]);
      for (int i = 0; i < 16; ++i) S[(row0 + i) * 64 + vcol] *= scale;
      __syncthreads();

      for (int j = 0; j < NH; ++j) {
        if (tid < 64) {
          const size_t base = ((size_t)b * Tt + tok) * (size_t)(Hh * HK * NH);
          kv[tid] = k[base + (size_t)(h * HK + tid) * NH + j];
          dv[tid] = v[base + (size_t)(h * HK + tid) * NH + j];
        }
        if (tid == 0) scl[0] = bet[((size_t)b * Tt + tok) * (Hh * NH) + h * NH + j];
        __syncthreads();

        // kS[v] = sum_k kv[k] * S[k][v]
        float p = 0.f;
        for (int i = 0; i < 16; ++i) p += kv[row0 + i] * S[(row0 + i) * 64 + vcol];
        red[tid] = p;
        __syncthreads();
        if (tid < 64) {
          const float ks = red[tid] + red[tid + 64] + red[tid + 128] + red[tid + 192];
          dv[tid] = dv[tid] - ks;    // d = v - kS
        }
        __syncthreads();

        const float bj = scl[0];
        const float dvv = dv[vcol];
        for (int i = 0; i < 16; ++i)
          S[(row0 + i) * 64 + vcol] += bj * kv[row0 + i] * dvv;
        __syncthreads();
      }

      // o[v] = sum_k q[k] * S[k][v]
      if (tid < 64) kv[tid] = q[((size_t)b * Tt + tok) * (Hh * HK) + h * HK + tid];
      __syncthreads();
      float p = 0.f;
      for (int i = 0; i < 16; ++i) p += kv[row0 + i] * S[(row0 + i) * 64 + vcol];
      red[tid] = p;
      __syncthreads();
      if (tid < 64) {
        const float o = red[tid] + red[tid + 64] + red[tid + 128] + red[tid + 192];
        outs[((size_t)b * Tt + tok) * Dd + h * HK + tid] = o;
      }
      __syncthreads();
    }

    // level merge: lev = trailing zeros of (c+1)
    const int lev = __ffs(c + 1) - 1;
    for (int i = 0; i < 16; ++i) {
      const int e = (row0 + i) * 64 + vcol;
      float acc = S[e];
      for (int l = 0; l < lev; ++l) {
        float* st = states + ((size_t)(b * LV + l) * Hh + h) * 4096;
        acc += st[e];
        st[e] = 0.f;
      }
      states[((size_t)(b * LV + lev) * Hh + h) * 4096 + e] = acc;
    }
    __syncthreads();
  }
}

// ---------------------------------------------------------------------------
// GroupNorm stats per (b,h) over (t, v): mean and rstd
// ---------------------------------------------------------------------------
__global__ __launch_bounds__(256)
void gn_stats_kernel(const float* __restrict__ outs, float* __restrict__ gnbuf) {
  const int bh = blockIdx.x;
  const int b = bh / Hh;
  const int h = bh % Hh;
  const int tid = threadIdx.x;
  float s = 0.f, s2 = 0.f;
  for (int idx = tid; idx < Tt * HK; idx += 256) {
    const int t = idx >> 6, vv = idx & 63;
    const float x = outs[((size_t)b * Tt + t) * Dd + h * HK + vv];
    s += x;
    s2 += x * x;
  }
  __shared__ float rs[256], rs2[256];
  rs[tid] = s;
  rs2[tid] = s2;
  __syncthreads();
  for (int off = 128; off > 0; off >>= 1) {
    if (tid < off) { rs[tid] += rs[tid + off]; rs2[tid] += rs2[tid + off]; }
    __syncthreads();
  }
  if (tid == 0) {
    const float inv = 1.f / (float)(Tt * HK);
    const float mean = rs[0] * inv;
    const float var = rs2[0] * inv - mean * mean;
    gnbuf[bh * 2]     = mean;
    gnbuf[bh * 2 + 1] = rsqrtf(var + 1e-5f);
  }
}

// ---------------------------------------------------------------------------
// y = ((outs - mean)*rstd*gamma + beta) * sigmoid(o_gate)  -> bf16
// ---------------------------------------------------------------------------
__global__ void gate_norm_kernel(const float* __restrict__ outs, const float* __restrict__ ogate,
                                 const float* __restrict__ gnbuf, const float* __restrict__ gamma,
                                 const float* __restrict__ gbeta, __bf16* __restrict__ y) {
  const size_t total = (size_t)MTOK * Dd;
  size_t i = (size_t)blockIdx.x * blockDim.x + threadIdx.x;
  if (i >= total) return;
  const int ch = (int)(i % Dd);
  const size_t bt = i / Dd;
  const int b = (int)(bt / Tt);
  const int h = ch / HK;
  const float mean = gnbuf[(b * Hh + h) * 2];
  const float rstd = gnbuf[(b * Hh + h) * 2 + 1];
  const float val = (outs[i] - mean) * rstd * gamma[ch] + gbeta[ch];
  const float gate = 1.f / (1.f + __expf(-ogate[i]));
  y[i] = f2bf(val * gate);
}

// ---------------------------------------------------------------------------
// Host launcher
// ---------------------------------------------------------------------------
extern "C" void kernel_launch(void* const* d_in, const int* in_sizes, int n_in,
                              void* d_out, int out_size, void* d_ws, size_t ws_size,
                              hipStream_t stream) {
  (void)in_sizes; (void)n_in; (void)out_size; (void)ws_size;
  const float* x     = (const float*)d_in[0];
  // d_in[1] cos, d_in[2] sin: unused by the reference forward
  const float* Wq    = (const float*)d_in[3];
  const float* Wk    = (const float*)d_in[4];
  const float* Wv    = (const float*)d_in[5];
  const float* Wb    = (const float*)d_in[6];
  const float* Wa    = (const float*)d_in[7];
  const float* Wg    = (const float*)d_in[8];
  const float* Wo    = (const float*)d_in[9];
  const float* Wl    = (const float*)d_in[10];
  const float* Lp    = (const float*)d_in[11];
  const float* gamma = (const float*)d_in[12];
  const float* gbeta = (const float*)d_in[13];
  float* out = (float*)d_out;

  char* wsp = (char*)d_ws;
  auto alloc = [&](size_t bytes) -> void* {
    void* p = (void*)wsp;
    wsp += (bytes + 255) & ~(size_t)255;
    return p;
  };

  const size_t nTokD = (size_t)MTOK * Dd;      // 4096*1024
  // bf16 staging buffers
  __bf16* xb   = (__bf16*)alloc(nTokD * 2);
  __bf16* wqb  = (__bf16*)alloc((size_t)1024 * 1024 * 2);
  __bf16* wkb  = (__bf16*)alloc((size_t)3072 * 1024 * 2);
  __bf16* wvb  = (__bf16*)alloc((size_t)3072 * 1024 * 2);
  __bf16* wbb  = (__bf16*)alloc((size_t)48 * 1024 * 2);
  __bf16* wab  = (__bf16*)alloc((size_t)16 * 1024 * 2);
  __bf16* wgb  = (__bf16*)alloc((size_t)1024 * 1024 * 2);
  __bf16* wlb  = (__bf16*)alloc((size_t)256 * 1024 * 2);
  __bf16* wob  = (__bf16*)alloc((size_t)1024 * 1024 * 2);
  __bf16* yb   = (__bf16*)alloc(nTokD * 2);
  // fp32 intermediates
  float* pq   = (float*)alloc((size_t)MTOK * 1024 * 4);
  float* pk   = (float*)alloc((size_t)MTOK * 3072 * 4);
  float* pv   = (float*)alloc((size_t)MTOK * 3072 * 4);
  float* pb   = (float*)alloc((size_t)MTOK * 48 * 4);
  float* pg   = (float*)alloc((size_t)MTOK * 16 * 4);
  float* pog  = (float*)alloc((size_t)MTOK * 1024 * 4);
  float* pdl  = (float*)alloc((size_t)MTOK * 256 * 4);
  float* lwc  = (float*)alloc((size_t)NCHUNK * Bc * Hh * LV * 4);
  float* st   = (float*)alloc((size_t)Bc * LV * Hh * 4096 * 4);
  float* outs = (float*)alloc(nTokD * 4);
  float* gnb  = (float*)alloc((size_t)Bc * Hh * 2 * 4);

  auto cvt = [&](const float* s, __bf16* d, size_t n) {
    cvt_bf16_kernel<<<4096, 256, 0, stream>>>(s, d, n);
  };
  cvt(x,  xb,  nTokD);
  cvt(Wq, wqb, (size_t)1024 * 1024);
  cvt(Wk, wkb, (size_t)3072 * 1024);
  cvt(Wv, wvb, (size_t)3072 * 1024);
  cvt(Wb, wbb, (size_t)48 * 1024);
  cvt(Wa, wab, (size_t)16 * 1024);
  cvt(Wg, wgb, (size_t)1024 * 1024);
  cvt(Wl, wlb, (size_t)256 * 1024);
  cvt(Wo, wob, (size_t)1024 * 1024);

  auto gemm = [&](const __bf16* A, const __bf16* W, float* C, int N) {
    dim3 grid((N + TN - 1) / TN, MTOK / TM);
    gemm_bf16_kernel<<<grid, 256, 0, stream>>>(A, W, C, MTOK, N, Dd);
  };
  gemm(xb, wqb, pq,  1024);
  gemm(xb, wkb, pk,  3072);
  gemm(xb, wvb, pv,  3072);
  gemm(xb, wbb, pb,  48);
  gemm(xb, wab, pg,  16);
  gemm(xb, wgb, pog, 1024);
  gemm(xb, wlb, pdl, 256);

  auto act = [&](float* p, size_t n, int mode) {
    act_kernel<<<(unsigned)((n + 255) / 256), 256, 0, stream>>>(p, n, mode);
  };
  act(pk, (size_t)MTOK * 3072, 0);   // silu
  act(pv, (size_t)MTOK * 3072, 0);   // silu
  act(pb, (size_t)MTOK * 48, 1);     // sigmoid
  act(pg, (size_t)MTOK * 16, 2);     // log_sigmoid

  lwc_kernel<<<NCHUNK * Bc * Hh, 64, 0, stream>>>(pdl, Lp, lwc);

  recurrence_kernel<<<Bc * Hh, 256, 0, stream>>>(pq, pk, pv, pb, pg, lwc, st, outs);

  gn_stats_kernel<<<Bc * Hh, 256, 0, stream>>>(outs, gnb);

  gate_norm_kernel<<<(unsigned)((nTokD + 255) / 256), 256, 0, stream>>>(
      outs, pog, gnb, gamma, gbeta, yb);

  gemm(yb, wob, out, 1024);
}
